// AutoCorrelation_63196148794095
// MI455X (gfx1250) — matrix operand order repaired and checked by reading, not compiled, not run on hardware
//
#include <hip/hip_runtime.h>
#include <hip/hip_bf16.h>
#include <math.h>

#define NB   16
#define NL   2048
#define NC   512      // H*E = 8*64, contiguous in [B,L,H,E]
#define TOPK 7
#define RS   136      // LDS row stride in bf16 elements (272 B, pads banks)

typedef __attribute__((ext_vector_type(16))) __bf16 v16bf;
typedef __attribute__((ext_vector_type(8)))  __bf16 v8bf;
typedef __attribute__((ext_vector_type(8)))  float  v8f;

__device__ __forceinline__ unsigned short f2bf(float x) {
    unsigned int u = __float_as_uint(x);
    u = u + 0x7FFFu + ((u >> 16) & 1u);          // round-to-nearest-even
    return (unsigned short)(u >> 16);
}

// ---------------- Kernel 0: f32 -> bf16 pack of Q and K, zero M ----------------
__global__ __launch_bounds__(256)
void k_pack(const float4* __restrict__ q4, const float4* __restrict__ k4,
            ushort4* __restrict__ qb, ushort4* __restrict__ kb,
            float* __restrict__ Mbuf) {
    int i = blockIdx.x * 256 + threadIdx.x;      // over 16M/4 = 4M
    float4 a = q4[i];
    float4 b = k4[i];
    qb[i] = make_ushort4(f2bf(a.x), f2bf(a.y), f2bf(a.z), f2bf(a.w));
    kb[i] = make_ushort4(f2bf(b.x), f2bf(b.y), f2bf(b.z), f2bf(b.w));
    if (i < NB * NL) Mbuf[i] = 0.0f;
}

// ---------------- Kernel 1: banded WMMA correlation, LDS sliding window --------
// Block = 8 waves = bands d0..d0+7 of one batch. At T-step T, wave w consumes
// B-tile S = T-d0-w: the block's window slides by one tile per step, so each
// step stages exactly one new A tile + one new B tile in LDS (shared 8 ways).
__global__ __launch_bounds__(256)
void k_corr(const unsigned short* __restrict__ Qb,
            const unsigned short* __restrict__ Kb,
            float* __restrict__ Mbuf) {
    __shared__ __align__(16) unsigned short Abuf[2][16 * RS];  // double-buffered A(T)
    __shared__ __align__(16) unsigned short Bbuf[9][16 * RS];  // 9-slot B ring
    __shared__ float tile[8][16][17];

    const int tid  = threadIdx.x;
    const int lane = tid & 31;
    const int wav  = tid >> 5;
    const int b    = blockIdx.x >> 4;            // 0..15
    const int d0   = (blockIdx.x & 15) * 8;      // first band of this block
    const int dlt  = d0 + wav;                   // this wave's band

    const int mrow  = lane & 15;                 // A row / B col / C col
    const int ahalf = lane >> 4;                 // 0 or 1

    const int lrow = tid >> 4;                   // cooperative loader: row 0..15
    const int lseg = tid & 15;                   // 16B segment 0..15
    const int stoff = lrow * RS + lseg * 8;      // LDS store offset (elements)

    const __bf16* Q = (const __bf16*)Qb + (size_t)b * (NL * NC);
    const __bf16* K = (const __bf16*)Kb + (size_t)b * (NL * NC);

    v8f acc = {};
    for (int cc = 0; cc < 4; ++cc) {             // channel chunks of 128
        const int cbase = cc * 128;
        const int gcol  = cbase + lseg * 8;

        // ---- refill window for T=0: A(0) + B tiles S = -d0-7 .. -d0 ----
        *(uint4*)(&Abuf[0][stoff]) =
            *(const uint4*)(Q + (size_t)(lrow) * NC + gcol);
#pragma unroll
        for (int j = 0; j < 8; ++j) {
            const int Su   = -d0 - j;            // unwrapped tile index
            const int srow = Su & 127;
            const int slot = (Su + 135) % 9;
            *(uint4*)(&Bbuf[slot][stoff]) =
                *(const uint4*)(K + (size_t)(srow * 16 + lrow) * NC + gcol);
        }
        __syncthreads();

        for (int T = 0; T < 128; ++T) {
            // prefetch next step's A + B tile into registers (overlaps compute)
            uint4 pa, pb;
            const int Tn = T + 1;
            if (Tn < 128) {
                pa = *(const uint4*)(Q + (size_t)(Tn * 16 + lrow) * NC + gcol);
                const int Su = Tn - d0;
                pb = *(const uint4*)(K + (size_t)((Su & 127) * 16 + lrow) * NC + gcol);
            }

            // compute: 4 WMMAs over this 128-channel chunk
            const unsigned short* Ab = &Abuf[T & 1][mrow * RS];
            const unsigned short* Bb = &Bbuf[(T - d0 - wav + 135) % 9][mrow * RS];
#pragma unroll
            for (int ch = 0; ch < 4; ++ch) {
                const int aoff = ch * 32 + ahalf * 8;
                v8bf alo = *(const v8bf*)((const __bf16*)Ab + aoff);
                v8bf ahi = *(const v8bf*)((const __bf16*)Ab + aoff + 16);
                v16bf amat = __builtin_shufflevector(alo, ahi,
                    0, 1, 2, 3, 4, 5, 6, 7, 8, 9, 10, 11, 12, 13, 14, 15);
                const int boff = ch * 32 + ahalf * 8;
                v8bf blo = *(const v8bf*)((const __bf16*)Bb + boff);
                v8bf bhi = *(const v8bf*)((const __bf16*)Bb + boff + 16);
                v16bf bmat = __builtin_shufflevector(blo, bhi,
                    0, 1, 2, 3, 4, 5, 6, 7, 8, 9, 10, 11, 12, 13, 14, 15);
                acc = __builtin_amdgcn_wmma_f32_16x16x32_bf16(
                    false, amat, false, bmat, (short)0, acc, false, false);
            }

            // stage prefetched tiles; targets are never live during step T
            if (Tn < 128) {
                *(uint4*)(&Abuf[Tn & 1][stoff]) = pa;
                *(uint4*)(&Bbuf[(Tn - d0 + 135) % 9][stoff]) = pb;
            }
            __syncthreads();
        }
    }

    // C layout: lanes 0-15 -> M=r, lanes 16-31 -> M=8+r ; N = lane & 15
#pragma unroll
    for (int r = 0; r < 8; ++r)
        tile[wav][r + ahalf * 8][mrow] = acc[r];
    __syncthreads();

    if (lane < 31) {
        const int d   = lane - 15;               // diagonal m-n in [-15,15]
        const int mlo = d > 0 ? d : 0;
        const int mhi = d > 0 ? 15 : 15 + d;
        float s = 0.0f;
        for (int m = mlo; m <= mhi; ++m) s += tile[wav][m][m - d];
        const int tau = (16 * dlt + d) & (NL - 1);
        atomicAdd(&Mbuf[b * NL + tau], s);
    }
}

// ---------------- Kernel 2: batch-avg top-7 + per-batch softmax ----------------
__global__ __launch_bounds__(256)
void k_topk(const float* __restrict__ Mbuf, int* __restrict__ topk,
            float* __restrict__ wgt) {
    __shared__ float s[NL];
    __shared__ int   sidx[TOPK];
    const int tid = threadIdx.x;

    for (int t = tid; t < NL; t += 256) {
        float v = 0.0f;
        for (int b = 0; b < NB; ++b) v += Mbuf[b * NL + t];
        s[t] = v;                                // monotone in batch-mean
    }
    __syncthreads();

    if (tid == 0) {
        for (int k = 0; k < TOPK; ++k) {
            float best = -3.4e38f; int bi = 0;
            for (int t = 0; t < NL; ++t)
                if (s[t] > best) { best = s[t]; bi = t; }
            sidx[k] = bi; topk[k] = bi; s[bi] = -3.4e38f;
        }
    }
    __syncthreads();

    if (tid < NB) {
        const int b = tid;
        float v[TOPK], mx = -3.4e38f;
        for (int k = 0; k < TOPK; ++k) {
            v[k] = Mbuf[b * NL + sidx[k]] * (1.0f / 512.0f);   // mean over H*E
            mx = fmaxf(mx, v[k]);
        }
        float sum = 0.0f;
        for (int k = 0; k < TOPK; ++k) { v[k] = expf(v[k] - mx); sum += v[k]; }
        const float inv = 1.0f / sum;
        for (int k = 0; k < TOPK; ++k) wgt[b * TOPK + k] = v[k] * inv;
    }
}

// ---------------- Kernel 3: weighted delay aggregation ----------------
// out[b,l,c] = sum_k w[b,k] * V[b,(l+d_k)%L, c]
__global__ __launch_bounds__(256)
void k_agg(const float4* __restrict__ V4, const int* __restrict__ topk,
           const float* __restrict__ wgt, float4* __restrict__ out4) {
    const int tid = blockIdx.x * 256 + threadIdx.x;  // over 16*2048*128
    const int c4 = tid & 127;
    const int l  = (tid >> 7) & (NL - 1);
    const int b  = tid >> 18;

    float4 r = make_float4(0.f, 0.f, 0.f, 0.f);
#pragma unroll
    for (int k = 0; k < TOPK; ++k) {
        const int   d  = topk[k];
        const float w  = wgt[b * TOPK + k];
        const int   ls = (l + d) & (NL - 1);
        const float4 v = V4[(size_t)(b * NL + ls) * 128 + c4];
        r.x += w * v.x; r.y += w * v.y; r.z += w * v.z; r.w += w * v.w;
    }
    out4[tid] = r;
}

extern "C" void kernel_launch(void* const* d_in, const int* in_sizes, int n_in,
                              void* d_out, int out_size, void* d_ws, size_t ws_size,
                              hipStream_t stream) {
    const float* q = (const float*)d_in[0];
    const float* k = (const float*)d_in[1];
    const float* v = (const float*)d_in[2];
    float* out = (float*)d_out;

    char* ws = (char*)d_ws;
    unsigned short* Qb   = (unsigned short*)(ws);                 // 32 MiB
    unsigned short* Kb   = (unsigned short*)(ws + 33554432);      // 32 MiB
    float*          Mbuf = (float*)(ws + 67108864);               // 128 KiB
    int*            topk = (int*)(ws + 67239936);
    float*          wgt  = (float*)(ws + 67239936 + 64);

    // 16M elements / 4 per thread = 4M threads
    k_pack<<<16384, 256, 0, stream>>>((const float4*)q, (const float4*)k,
                                      (ushort4*)Qb, (ushort4*)Kb, Mbuf);
    // 256 blocks x 8 waves = 16 batches x 128 bands
    k_corr<<<256, 256, 0, stream>>>(Qb, Kb, Mbuf);
    k_topk<<<1, 256, 0, stream>>>(Mbuf, topk, wgt);
    // 16*2048*128 float4 outputs
    k_agg<<<16384, 256, 0, stream>>>((const float4*)v, topk, wgt, (float4*)out);
}
